// SSTransformer_17652315586810
// MI455X (gfx1250) — compile-verified, matrix operand order checked by reading, not run on hardware
//
#include <hip/hip_runtime.h>
#include <hip/hip_bf16.h>
#include <math.h>
#include <stdint.h>

typedef __attribute__((ext_vector_type(16))) _Float16 v16h;
typedef __attribute__((ext_vector_type(8)))  float    v8f;
typedef __attribute__((ext_vector_type(4)))  unsigned int u32x4;
typedef __attribute__((ext_vector_type(8)))  int          i32x8;
typedef __attribute__((ext_vector_type(4)))  int          i32x4;

// Problem constants (fixed by the reference harness)
constexpr int  Bn = 4;
constexpr int  Cc = 64;           // channels
constexpr int  Hh = 256, Ww = 256;
constexpr int  Nn = Hh * Ww;      // 65536 tokens
constexpr int  HEADS = 8;
constexpr int  CH = Cc / HEADS;   // 8

#if defined(__HIP_DEVICE_COMPILE__) && defined(__has_builtin)
#  if __has_builtin(__builtin_amdgcn_tensor_load_to_lds) && __has_builtin(__builtin_amdgcn_s_wait_tensorcnt)
#    define HAVE_TDM 1
#  endif
#endif
#ifndef HAVE_TDM
#  define HAVE_TDM 0
#endif

// K index inside one 16x32 f16 WMMA operand for element e (0..15) of lane L.
// lane<16: elems 0-7 -> K0-7, elems 8-15 -> K16-23 ; lane>=16: +8.
__device__ __forceinline__ int wmma_kidx(int e, int lane) {
    return (e & 7) + ((e >> 3) << 4) + ((lane >> 4) << 3);
}

__device__ __forceinline__ float wave_reduce_add(float v) {
    #pragma unroll
    for (int m = 16; m >= 1; m >>= 1) v += __shfl_xor(v, m, 32);
    return v;
}

// ---------------------------------------------------------------------------
// Kernel 1: qkv 1x1 conv as GEMM  qkv[b,oc,n] = sum_c W[oc,c] * x[b,c,n]
// Weights (192x64 f32, 48KB) staged into LDS once per block via the Tensor
// Data Mover with a 1-DWORD pad every 64 DWORDs (row stride 65 -> conflict-
// free A-operand reads). One wave per 16-pixel tile sweeps all 12 oc-tiles.
// ---------------------------------------------------------------------------
__global__ __launch_bounds__(256) void k_qkv_gemm(const float* __restrict__ x,
                                                  const float* __restrict__ w,
                                                  float* __restrict__ qkv) {
    __shared__ float wlds[192 * 65];       // 12480 floats, padded rows

#if HAVE_TDM
    if (threadIdx.x < 32) {                // one wave issues the TDM descriptor
        const unsigned lds_off = (unsigned)(uintptr_t)(void*)wlds;  // low 32 bits = LDS offset
        const unsigned long long ga = (unsigned long long)(uintptr_t)w;
        // D# group0: count=1, lds_addr, 57-bit global addr, type=2 (bits 127:126)
        u32x4 g0 = { 1u, lds_off, (unsigned)ga,
                     (unsigned)((ga >> 32) & 0x01FFFFFFu) | (2u << 30) };
        // D# group1: data_size=4B (code 2), pad_enable, pad_interval=64 DW (code 5),
        // pad_amount=1 DW (code 0); tensor_dim0=tile_dim0=12288, dim1=1, stride0=12288.
        i32x8 g1 = { (int)0x01520000,      // mask0 | data_size | pad ctl
                     (int)0x30000000,      // tensor_dim0[15:0] << 16
                     (int)0x00010000,      // tensor_dim0[31:16] | tensor_dim1[15:0]<<16
                     (int)0x30000000,      // tensor_dim1[31:16] | tile_dim0 << 16
                     (int)0x00000001,      // tile_dim1=1, tile_dim2=0
                     (int)0x00003000,      // tensor_dim0_stride[31:0] = 12288
                     0, 0 };
        i32x4 z4 = { 0, 0, 0, 0 };
#  if __clang_major__ >= 23
        i32x8 z8 = { 0, 0, 0, 0, 0, 0, 0, 0 };
        __builtin_amdgcn_tensor_load_to_lds(g0, g1, z4, z4, z8, 0);
#  else
        __builtin_amdgcn_tensor_load_to_lds(g0, g1, z4, z4, 0);
#  endif
        __builtin_amdgcn_s_wait_tensorcnt(0);
    }
    __syncthreads();
#else
    for (int i = threadIdx.x; i < 192 * 64; i += 256) wlds[i + (i >> 6)] = w[i];
    __syncthreads();
#endif

    const int lane = threadIdx.x & 31;
    const int wv   = threadIdx.x >> 5;
    const int tile = blockIdx.x * 8 + wv;          // 16384 tiles total
    const int tilesPerBatch = Nn / 16;             // 4096
    const int b  = tile / tilesPerBatch;
    const int pt = tile % tilesPerBatch;
    const int p  = pt * 16 + (lane & 15);

    const float* xb = x + (long)b * Cc * Nn;
    v16h b_lo, b_hi;
    #pragma unroll
    for (int e = 0; e < 16; ++e) {
        const int kk = wmma_kidx(e, lane);
        b_lo[e] = (_Float16)xb[(long)kk * Nn + p];
        b_hi[e] = (_Float16)xb[(long)(kk + 32) * Nn + p];
    }

    float* outb = qkv + (long)b * (3 * Cc) * Nn;
    for (int ot = 0; ot < 12; ++ot) {
        const int m = ot * 16 + (lane & 15);
        v16h a_lo, a_hi;
        #pragma unroll
        for (int e = 0; e < 16; ++e) {
            const int kk = wmma_kidx(e, lane);
            a_lo[e] = (_Float16)wlds[m * 65 + kk];
            a_hi[e] = (_Float16)wlds[m * 65 + kk + 32];
        }
        v8f c = {};
        c = __builtin_amdgcn_wmma_f32_16x16x32_f16(false, a_lo, false, b_lo, (short)0, c, false, false);
        c = __builtin_amdgcn_wmma_f32_16x16x32_f16(false, a_hi, false, b_hi, (short)0, c, false, false);
        #pragma unroll
        for (int r = 0; r < 8; ++r) {
            const int oc = ot * 16 + r + ((lane >> 4) << 3);
            outb[(long)oc * Nn + p] = c[r];
        }
    }
}

// ---------------------------------------------------------------------------
// Kernel 2/6/7: depthwise 3x3 SAME conv, 4 pixels per thread (float4 I/O).
// mode 0 = plain, 1 = + exact GELU, 2 = accumulate into out (out += conv).
// ---------------------------------------------------------------------------
__global__ __launch_bounds__(256) void k_dw3x3(const float* __restrict__ in, long inBStride,
                                               const float* __restrict__ w9,
                                               float* __restrict__ out, long outBStride,
                                               int Cch, int mode) {
    const long id     = (long)blockIdx.x * blockDim.x + threadIdx.x;   // quad id
    const long totalQ = (long)Bn * Cch * Nn / 4;
    if (id >= totalQ) return;
    const int wq = Ww / 4;
    const int xq = (int)(id % wq);
    const int y  = (int)((id / wq) % Hh);
    const int c  = (int)((id / ((long)Nn / 4)) % Cch);
    const int b  = (int)(id / ((long)Cch * Nn / 4));
    const int x0 = xq * 4;

    const float* ip = in + (long)b * inBStride + (long)c * Nn;
    const float* wp = w9 + c * 9;

    float r[3][6];
    #pragma unroll
    for (int dy = 0; dy < 3; ++dy) {
        const int yy = y + dy - 1;
        if (yy < 0 || yy >= Hh) {
            #pragma unroll
            for (int i = 0; i < 6; ++i) r[dy][i] = 0.f;
            continue;
        }
        const float* rp = ip + (long)yy * Ww + x0;
        const float4 m = *(const float4*)rp;        // 16B aligned
        r[dy][1] = m.x; r[dy][2] = m.y; r[dy][3] = m.z; r[dy][4] = m.w;
        r[dy][0] = (x0 > 0)       ? rp[-1] : 0.f;
        r[dy][5] = (x0 + 4 < Ww)  ? rp[4]  : 0.f;
    }

    float o[4];
    #pragma unroll
    for (int j = 0; j < 4; ++j) {
        float s = 0.f;
        #pragma unroll
        for (int dy = 0; dy < 3; ++dy)
            #pragma unroll
            for (int dx = 0; dx < 3; ++dx)
                s = fmaf(wp[dy * 3 + dx], r[dy][j + dx], s);
        o[j] = s;
    }
    if (mode == 1) {
        #pragma unroll
        for (int j = 0; j < 4; ++j)
            o[j] = 0.5f * o[j] * (1.f + erff(o[j] * 0.70710678118654752f));  // exact GELU
    }
    float* op = out + (long)b * outBStride + (long)c * Nn + (long)y * Ww + x0;
    float4 res = { o[0], o[1], o[2], o[3] };
    if (mode == 2) {
        const float4 prev = *(const float4*)op;
        res.x += prev.x; res.y += prev.y; res.z += prev.z; res.w += prev.w;
    }
    *(float4*)op = res;
}

// ---------------------------------------------------------------------------
// Kernel 3: per-(b,head) reductions over N: Gram S_qk (8x8), sum q^2, sum k^2.
// red layout per bh: [64 Sqk][8 nq2][8 nk2]
// ---------------------------------------------------------------------------
__global__ __launch_bounds__(256) void k_attn_reduce(const float* __restrict__ qkvdw,
                                                     float* __restrict__ red) {
    const int bh = blockIdx.y;                    // 0..31
    const int b  = bh >> 3, hd = bh & 7;
    const float* qp = qkvdw + ((long)(b * 192 + hd * CH)) * Nn;
    const float* kp = qkvdw + ((long)(b * 192 + 64 + hd * CH)) * Nn;

    float acc[8][8], q2[8], k2[8];
    #pragma unroll
    for (int i = 0; i < 8; ++i) {
        q2[i] = 0.f; k2[i] = 0.f;
        #pragma unroll
        for (int j = 0; j < 8; ++j) acc[i][j] = 0.f;
    }

    const int nPer = Nn / gridDim.x;              // 1024 with gridDim.x = 64
    const int n0   = blockIdx.x * nPer;
    for (int n = n0 + threadIdx.x; n < n0 + nPer; n += 256) {
        float qv[8], kv[8];
        #pragma unroll
        for (int c = 0; c < 8; ++c) { qv[c] = qp[(long)c * Nn + n]; kv[c] = kp[(long)c * Nn + n]; }
        #pragma unroll
        for (int c = 0; c < 8; ++c) {
            q2[c] = fmaf(qv[c], qv[c], q2[c]);
            k2[c] = fmaf(kv[c], kv[c], k2[c]);
            #pragma unroll
            for (int d = 0; d < 8; ++d) acc[c][d] = fmaf(qv[c], kv[d], acc[c][d]);
        }
    }

    __shared__ float sred[80];
    if (threadIdx.x < 80) sred[threadIdx.x] = 0.f;
    __syncthreads();
    const int lane = threadIdx.x & 31;
    #pragma unroll
    for (int c = 0; c < 8; ++c) {
        #pragma unroll
        for (int d = 0; d < 8; ++d) {
            const float v = wave_reduce_add(acc[c][d]);
            if (lane == 0) atomicAdd(&sred[c * 8 + d], v);
        }
        const float vq = wave_reduce_add(q2[c]);
        if (lane == 0) atomicAdd(&sred[64 + c], vq);
        const float vk = wave_reduce_add(k2[c]);
        if (lane == 0) atomicAdd(&sred[72 + c], vk);
    }
    __syncthreads();
    if (threadIdx.x < 80) atomicAdd(&red[(long)bh * 80 + threadIdx.x], sred[threadIdx.x]);
}

// ---------------------------------------------------------------------------
// Kernel 4: normalize Gram, softmax + rel_bias, fold with proj_w into the
// per-batch 64x64 matrix M (out = M @ v + proj_b).
// ---------------------------------------------------------------------------
__global__ __launch_bounds__(256) void k_finalize(const float* __restrict__ red,
                                                  const float* __restrict__ proj_w,
                                                  const float* __restrict__ temperature,
                                                  const float* __restrict__ rel_bias,
                                                  float* __restrict__ Mmat) {
    __shared__ float sattn[8][8][8];  // [head][row c][col d]
    const int b   = blockIdx.x;
    const int tid = threadIdx.x;
    if (tid < 64) {
        const int hd = tid >> 3, cc = tid & 7;
        const float* r = red + ((long)(b * 8 + hd)) * 80;
        const float nq = fmaxf(sqrtf(r[64 + cc]), 1e-12f);
        float vrow[8];
        float mx = -1e30f;
        #pragma unroll
        for (int d = 0; d < 8; ++d) {
            const float nk = fmaxf(sqrtf(r[72 + d]), 1e-12f);
            vrow[d] = r[cc * 8 + d] / (nq * nk) * temperature[hd];
            mx = fmaxf(mx, vrow[d]);
        }
        float sum = 0.f;
        #pragma unroll
        for (int d = 0; d < 8; ++d) { vrow[d] = expf(vrow[d] - mx); sum += vrow[d]; }
        const float inv = 1.f / sum;
        #pragma unroll
        for (int d = 0; d < 8; ++d)
            sattn[hd][cc][d] = vrow[d] * inv + rel_bias[cc - d + CH - 1];
    }
    __syncthreads();
    for (int i = tid; i < 4096; i += 256) {
        const int o = i >> 6, cp = i & 63;
        const int h = cp >> 3, d = cp & 7;
        float s = 0.f;
        #pragma unroll
        for (int j = 0; j < 8; ++j) s = fmaf(proj_w[o * 64 + h * 8 + j], sattn[h][j][d], s);
        Mmat[b * 4096 + i] = s;
    }
}

// ---------------------------------------------------------------------------
// Kernel 5: out[b,o,n] = sum_c M_b[o,c] * v[b,c,n] + proj_b[o]  (WMMA GEMM)
// ---------------------------------------------------------------------------
__global__ __launch_bounds__(256) void k_out_gemm(const float* __restrict__ qkvdw,
                                                  const float* __restrict__ Mmat,
                                                  const float* __restrict__ proj_b,
                                                  float* __restrict__ out) {
    const int lane = threadIdx.x & 31;
    const int wv   = threadIdx.x >> 5;
    const int tile = blockIdx.x * 8 + wv;
    const int tilesPerBatch = Nn / 16;
    const int b  = tile / tilesPerBatch;
    const int pt = tile % tilesPerBatch;
    const int p  = pt * 16 + (lane & 15);

    const float* vb = qkvdw + (long)b * 192 * Nn + (long)128 * Nn;   // v channels
    const float* Mb = Mmat + b * 4096;
    __builtin_prefetch(Mb, 0, 1);                  // global_prefetch_b8

    v16h b_lo, b_hi;
    #pragma unroll
    for (int e = 0; e < 16; ++e) {
        const int kk = wmma_kidx(e, lane);
        b_lo[e] = (_Float16)vb[(long)kk * Nn + p];
        b_hi[e] = (_Float16)vb[(long)(kk + 32) * Nn + p];
    }

    float* ob = out + (long)b * Cc * Nn;
    #pragma unroll
    for (int ot = 0; ot < 4; ++ot) {
        const int m = ot * 16 + (lane & 15);
        v16h a_lo, a_hi;
        #pragma unroll
        for (int e = 0; e < 16; ++e) {
            const int kk = wmma_kidx(e, lane);
            a_lo[e] = (_Float16)Mb[m * 64 + kk];
            a_hi[e] = (_Float16)Mb[m * 64 + kk + 32];
        }
        v8f c = {};
        c = __builtin_amdgcn_wmma_f32_16x16x32_f16(false, a_lo, false, b_lo, (short)0, c, false, false);
        c = __builtin_amdgcn_wmma_f32_16x16x32_f16(false, a_hi, false, b_hi, (short)0, c, false, false);
        #pragma unroll
        for (int r = 0; r < 8; ++r) {
            const int oc = ot * 16 + r + ((lane >> 4) << 3);
            ob[(long)oc * Nn + p] = c[r] + proj_b[oc];
        }
    }
}

__global__ void k_zero(float* p, int n) {
    const int i = blockIdx.x * blockDim.x + threadIdx.x;
    if (i < n) p[i] = 0.f;
}

// ---------------------------------------------------------------------------
extern "C" void kernel_launch(void* const* d_in, const int* in_sizes, int n_in,
                              void* d_out, int out_size, void* d_ws, size_t ws_size,
                              hipStream_t stream) {
    const float* x      = (const float*)d_in[0];
    const float* qkv_w  = (const float*)d_in[1];
    const float* dw_w   = (const float*)d_in[2];
    const float* proj_w = (const float*)d_in[3];
    const float* proj_b = (const float*)d_in[4];
    const float* pos1_w = (const float*)d_in[5];
    const float* pos2_w = (const float*)d_in[6];
    const float* temp   = (const float*)d_in[7];
    const float* relb   = (const float*)d_in[8];
    float* out = (float*)d_out;

    const size_t qkvElems = (size_t)Bn * 192 * Nn;     // 50.3M floats
    float* ws       = (float*)d_ws;
    float* qkv_raw  = ws;                    // 201 MB (reused as pe_t later)
    float* qkv_dw   = ws + qkvElems;         // 201 MB
    float* red      = qkv_dw + qkvElems;     // 2560 floats
    float* Mmat     = red + 2560;            // 16384 floats
    float* pe_t     = qkv_raw;               // reuse: qkv_raw dead after dwconv

    // zero the atomic-reduction region every call (graph replays)
    k_zero<<<(2560 + 255) / 256, 256, 0, stream>>>(red, 2560);

    // 1) qkv = W_qkv @ x  (WMMA, TDM-staged weights)
    k_qkv_gemm<<<(Bn * Nn / 16) / 8, 256, 0, stream>>>(x, qkv_w, qkv_raw);

    // 2) depthwise 3x3 on all 192 channels
    {
        const long totalQ = (long)Bn * 192 * Nn / 4;
        k_dw3x3<<<(int)((totalQ + 255) / 256), 256, 0, stream>>>(
            qkv_raw, (long)192 * Nn, dw_w, qkv_dw, (long)192 * Nn, 192, 0);
    }

    // 3) Gram/norm reductions per (b, head)
    k_attn_reduce<<<dim3(64, Bn * HEADS), 256, 0, stream>>>(qkv_dw, red);

    // 4) softmax + rel_bias, fold with proj -> per-batch 64x64 M
    k_finalize<<<Bn, 256, 0, stream>>>(red, proj_w, temp, relb, Mmat);

    // 5) out = M @ v + proj_b  (WMMA)
    k_out_gemm<<<(Bn * Nn / 16) / 8, 256, 0, stream>>>(qkv_dw, Mmat, proj_b, out);

    // 6) pe branch: t = GELU(dwconv(v, pos1)), then 7) out += dwconv(t, pos2)
    {
        const long totalQ = (long)Bn * Cc * Nn / 4;
        k_dw3x3<<<(int)((totalQ + 255) / 256), 256, 0, stream>>>(
            qkv_dw + (long)128 * Nn, (long)192 * Nn, pos1_w, pe_t, (long)Cc * Nn, Cc, 1);
        k_dw3x3<<<(int)((totalQ + 255) / 256), 256, 0, stream>>>(
            pe_t, (long)Cc * Nn, pos2_w, out, (long)Cc * Nn, Cc, 2);
    }
}
